// RegionGridCorrelation_16372415332370
// MI455X (gfx1250) — matrix operand order repaired and checked by reading, not compiled, not run on hardware
//
#include <hip/hip_runtime.h>
#include <hip/hip_bf16.h>

typedef __attribute__((ext_vector_type(16))) _Float16 v16h;
typedef __attribute__((ext_vector_type(8)))  float    v8f;

#define NBOX   463
#define NTOT   512
#define BATCH  4
#define HEADS  8
#define DK     64
#define DMODEL 512

// 1000^(-t/8), t = 0..7
static __device__ __constant__ float c_invfreq[8] = {
    1.0f, 0.421696503f, 0.177827941f, 0.0749894209f,
    0.0316227766f, 0.0133352143f, 0.00562341325f, 0.00237137371f};

// ---- WMMA tile loaders (16x16x32 f16, wave32, contiguous k = 16*(lane/16) + element in A and B) ----
// A fragment: lane m = l&15; elements 0..15 -> K (0..15)+ofs,
// ofs = 16 for lanes 16..31.  16 contiguous floats -> four 16B vector loads.
__device__ __forceinline__ v16h load_a_rm(const float* __restrict__ A, int lda, int m0, int k0) {
  const int lane = threadIdx.x & 31;
  const float* row = A + (size_t)(m0 + (lane & 15)) * lda + k0 + ((lane & 16) ? 16 : 0);
  const float4 f0 = ((const float4*)row)[0];
  const float4 f1 = ((const float4*)row)[1];
  const float4 f2 = ((const float4*)row)[2];
  const float4 f3 = ((const float4*)row)[3];
  v16h a;
  a[0]  = (_Float16)f0.x; a[1]  = (_Float16)f0.y; a[2]  = (_Float16)f0.z; a[3]  = (_Float16)f0.w;
  a[4]  = (_Float16)f1.x; a[5]  = (_Float16)f1.y; a[6]  = (_Float16)f1.z; a[7]  = (_Float16)f1.w;
  a[8]  = (_Float16)f2.x; a[9]  = (_Float16)f2.y; a[10] = (_Float16)f2.z; a[11] = (_Float16)f2.w;
  a[12] = (_Float16)f3.x; a[13] = (_Float16)f3.y; a[14] = (_Float16)f3.z; a[15] = (_Float16)f3.w;
  return a;
}

// B fragment (32x16, KxN) from row-major [K x N] matrix with row stride ldb.
// Lane n = l&15 holds K = i + (l>=16 ? 16 : 0): column walk, coalesced across lanes.
__device__ __forceinline__ v16h load_b_rm(const float* __restrict__ B, int ldb, int k0, int n0) {
  const int lane = threadIdx.x & 31;
  const float* col = B + (size_t)(k0 + ((lane & 16) ? 16 : 0)) * ldb + n0 + (lane & 15);
  v16h b;
#pragma unroll
  for (int i = 0; i < 16; ++i) b[i] = (_Float16)col[(size_t)i * ldb];
  return b;
}

// B fragment where memory holds the transpose: row-major [N x K], B[k][n] = Bt[n][k].
// 16 contiguous floats per lane -> four 16B vector loads.
__device__ __forceinline__ v16h load_b_tr(const float* __restrict__ Bt, int ldb, int n0, int k0) {
  const int lane = threadIdx.x & 31;
  const float* row = Bt + (size_t)(n0 + (lane & 15)) * ldb + k0 + ((lane & 16) ? 16 : 0);
  const float4 f0 = ((const float4*)row)[0];
  const float4 f1 = ((const float4*)row)[1];
  const float4 f2 = ((const float4*)row)[2];
  const float4 f3 = ((const float4*)row)[3];
  v16h b;
  b[0]  = (_Float16)f0.x; b[1]  = (_Float16)f0.y; b[2]  = (_Float16)f0.z; b[3]  = (_Float16)f0.w;
  b[4]  = (_Float16)f1.x; b[5]  = (_Float16)f1.y; b[6]  = (_Float16)f1.z; b[7]  = (_Float16)f1.w;
  b[8]  = (_Float16)f2.x; b[9]  = (_Float16)f2.y; b[10] = (_Float16)f2.z; b[11] = (_Float16)f2.w;
  b[12] = (_Float16)f3.x; b[13] = (_Float16)f3.y; b[14] = (_Float16)f3.z; b[15] = (_Float16)f3.w;
  return b;
}

// ---- Kernel 1: per-box geometry (cx, cy, w, h), boxes + grid cells ----
__global__ void boxes_kernel(const float* __restrict__ f_g, float* __restrict__ boxes) {
  int idx = blockIdx.x * blockDim.x + threadIdx.x;  // 0 .. 2047
  if (idx >= BATCH * NTOT) return;
  int n = idx & (NTOT - 1);
  float x0, y0, x1, y1;
  if (n < NBOX) {
    int b = idx >> 9;
    const float* p = f_g + ((size_t)b * NBOX + n) * 4;
    x0 = p[0]; y0 = p[1]; x1 = p[2]; y1 = p[3];
  } else {
    int g = n - NBOX;                 // 0..48
    float c1 = (float)(g / 7);        // row (torch quirk: feeds x)
    float c2 = (float)(g % 7);        // col (feeds y)
    x0 = c1 * (1.0f / 7.0f); y0 = c2 * (1.0f / 7.0f);
    x1 = (c1 + 1.0f) * (1.0f / 7.0f); y1 = (c2 + 1.0f) * (1.0f / 7.0f);
  }
  float* o = boxes + (size_t)idx * 4;
  o[0] = 0.5f * (x0 + x1);
  o[1] = 0.5f * (y0 + y1);
  o[2] = x1 - x0 + 1.0f;
  o[3] = y1 - y0 + 1.0f;
}

// ---- Kernel 2: log(clip(relu(emb . Wg + bg))) -> gb[b,h,q,k] ----
__global__ void geom_bias_kernel(const float* __restrict__ boxes, const float* __restrict__ Wg,
                                 const float* __restrict__ bg, float* __restrict__ gb) {
  __shared__ float sWg[HEADS * 64];
  for (int i = threadIdx.x; i < HEADS * 64; i += blockDim.x) sWg[i] = Wg[i];
  __syncthreads();
  int idx = blockIdx.x * blockDim.x + threadIdx.x;  // b*512*512 threads
  int k = idx & 511;
  int q = (idx >> 9) & 511;
  int b = idx >> 18;
  const float4 bq4 = ((const float4*)boxes)[b * NTOT + q];
  const float4 bk4 = ((const float4*)boxes)[b * NTOT + k];
  float pos[4];
  pos[0] = 100.0f * logf(fmaxf(fabsf((bq4.x - bk4.x) / bq4.z), 1e-3f));
  pos[1] = 100.0f * logf(fmaxf(fabsf((bq4.y - bk4.y) / bq4.w), 1e-3f));
  pos[2] = 100.0f * logf(bq4.z / bk4.z);
  pos[3] = 100.0f * logf(bq4.w / bk4.w);
  float acc[8];
#pragma unroll
  for (int h = 0; h < 8; ++h) acc[h] = bg[h];
#pragma unroll
  for (int p = 0; p < 4; ++p) {
#pragma unroll
    for (int t = 0; t < 8; ++t) {
      float m = pos[p] * c_invfreq[t];
      float s, c;
      sincosf(m, &s, &c);
      const int d = p * 8 + t;
#pragma unroll
      for (int h = 0; h < 8; ++h)
        acc[h] += s * sWg[h * 64 + d] + c * sWg[h * 64 + 32 + d];
    }
  }
  size_t base = (((size_t)b * 8) * 512 + q) * 512 + k;
#pragma unroll
  for (int h = 0; h < 8; ++h)
    gb[base + (size_t)h * 512 * 512] = logf(fmaxf(acc[h], 1e-6f));  // relu+clip+log
}

// ---- Kernel 3: X(2048x512) @ W(512x512) + bias -> Out[b,h,n,dk] (WMMA, 16x64 strip) ----
__global__ void proj_kernel(const float* __restrict__ X, const float* __restrict__ W,
                            const float* __restrict__ bias, float* __restrict__ Out) {
  const int wave = (blockIdx.x * blockDim.x + threadIdx.x) >> 5;  // 1024 strips
  const int lane = threadIdx.x & 31;
  const int tm = wave >> 3;   // 0..127 (M tiles of 16)
  const int ts = wave & 7;    // 0..7   (N strips of 64 == one head)
  v8f c0 = {}, c1 = {}, c2 = {}, c3 = {};
  for (int k0 = 0; k0 < DMODEL; k0 += 32) {
    v16h a = load_a_rm(X, DMODEL, tm * 16, k0);
    __builtin_prefetch(X + (size_t)(tm * 16 + (lane & 15)) * DMODEL + k0 + 32, 0, 1);
    v16h b0 = load_b_rm(W, DMODEL, k0, ts * 64);
    v16h b1 = load_b_rm(W, DMODEL, k0, ts * 64 + 16);
    v16h b2 = load_b_rm(W, DMODEL, k0, ts * 64 + 32);
    v16h b3 = load_b_rm(W, DMODEL, k0, ts * 64 + 48);
    c0 = __builtin_amdgcn_wmma_f32_16x16x32_f16(false, a, false, b0, (short)0, c0, false, false);
    c1 = __builtin_amdgcn_wmma_f32_16x16x32_f16(false, a, false, b1, (short)0, c1, false, false);
    c2 = __builtin_amdgcn_wmma_f32_16x16x32_f16(false, a, false, b2, (short)0, c2, false, false);
    c3 = __builtin_amdgcn_wmma_f32_16x16x32_f16(false, a, false, b3, (short)0, c3, false, false);
  }
  const int h = ts;  // strip == head
  v8f cc[4] = {c0, c1, c2, c3};
#pragma unroll
  for (int t = 0; t < 4; ++t) {
    const int j = ts * 64 + t * 16 + (lane & 15);
    const float bj = bias[j];
    const int d = t * 16 + (lane & 15);
#pragma unroll
    for (int r = 0; r < 8; ++r) {
      const int m = tm * 16 + r + ((lane & 16) ? 8 : 0);
      const int b = m >> 9, n = m & 511;
      Out[(((size_t)(b * 8 + h)) * 512 + n) * 64 + d] = cc[t][r] + bj;
    }
  }
}

// ---- Kernel 4: S[b,h,q,k] = (Q.K^T)/8 + gb (in place over gb) (WMMA, 16x64 strip) ----
__global__ void qk_bias_kernel(const float* __restrict__ Q, const float* __restrict__ K,
                               float* __restrict__ S) {
  const int wave = (blockIdx.x * blockDim.x + threadIdx.x) >> 5;  // 8192 strips
  const int lane = threadIdx.x & 31;
  const int bh = wave >> 8;
  const int rest = wave & 255;
  const int tm = rest >> 3;   // 0..31
  const int ts = rest & 7;    // 0..7 (key strip of 64)
  const float* q = Q + (size_t)bh * 512 * 64;
  const float* kk = K + (size_t)bh * 512 * 64;
  v8f c0 = {}, c1 = {}, c2 = {}, c3 = {};
  for (int k0 = 0; k0 < DK; k0 += 32) {
    v16h a = load_a_rm(q, DK, tm * 16, k0);
    v16h b0 = load_b_tr(kk, DK, ts * 64, k0);        // B[k][n] = Kmat[n][k]
    v16h b1 = load_b_tr(kk, DK, ts * 64 + 16, k0);
    v16h b2 = load_b_tr(kk, DK, ts * 64 + 32, k0);
    v16h b3 = load_b_tr(kk, DK, ts * 64 + 48, k0);
    c0 = __builtin_amdgcn_wmma_f32_16x16x32_f16(false, a, false, b0, (short)0, c0, false, false);
    c1 = __builtin_amdgcn_wmma_f32_16x16x32_f16(false, a, false, b1, (short)0, c1, false, false);
    c2 = __builtin_amdgcn_wmma_f32_16x16x32_f16(false, a, false, b2, (short)0, c2, false, false);
    c3 = __builtin_amdgcn_wmma_f32_16x16x32_f16(false, a, false, b3, (short)0, c3, false, false);
  }
  float* s = S + (size_t)bh * 512 * 512;
  v8f cc[4] = {c0, c1, c2, c3};
#pragma unroll
  for (int t = 0; t < 4; ++t) {
    const int col = ts * 64 + t * 16 + (lane & 15);
#pragma unroll
    for (int r = 0; r < 8; ++r) {
      const int row = tm * 16 + r + ((lane & 16) ? 8 : 0);
      size_t o = (size_t)row * 512 + col;
      s[o] = cc[t][r] * 0.125f + s[o];  // + log-geom bias already resident
    }
  }
}

// ---- Kernel 5: row softmax in place over S (one block per 512-wide row) ----
__global__ void softmax_kernel(float* __restrict__ S) {
  __shared__ float red[256];
  float* row = S + (size_t)blockIdx.x * 512;
  const int tid = threadIdx.x;
  float v0 = row[tid], v1 = row[tid + 256];
  red[tid] = fmaxf(v0, v1);
  __syncthreads();
  for (int s = 128; s > 0; s >>= 1) {
    if (tid < s) red[tid] = fmaxf(red[tid], red[tid + s]);
    __syncthreads();
  }
  const float M = red[0];
  __syncthreads();
  float e0 = __expf(v0 - M), e1 = __expf(v1 - M);
  red[tid] = e0 + e1;
  __syncthreads();
  for (int s = 128; s > 0; s >>= 1) {
    if (tid < s) red[tid] += red[tid + s];
    __syncthreads();
  }
  const float inv = 1.0f / red[0];
  row[tid] = e0 * inv;
  row[tid + 256] = e1 * inv;
}

// ---- Kernel 6: O = P(512x512) @ V(512x64) -> AO[b,q,h,d] (WMMA, 16x64 strip) ----
__global__ void pv_kernel(const float* __restrict__ P, const float* __restrict__ V,
                          float* __restrict__ AO) {
  const int wave = (blockIdx.x * blockDim.x + threadIdx.x) >> 5;  // 1024 strips
  const int lane = threadIdx.x & 31;
  const int bh = wave >> 5;
  const int tm = wave & 31;   // full 64-wide output covered by one strip
  const float* p = P + (size_t)bh * 512 * 512;
  const float* v = V + (size_t)bh * 512 * 64;
  v8f c0 = {}, c1 = {}, c2 = {}, c3 = {};
  for (int k0 = 0; k0 < 512; k0 += 32) {
    v16h a = load_a_rm(p, 512, tm * 16, k0);
    __builtin_prefetch(p + (size_t)(tm * 16 + (lane & 15)) * 512 + k0 + 32, 0, 1);
    v16h b0 = load_b_rm(v, 64, k0, 0);
    v16h b1 = load_b_rm(v, 64, k0, 16);
    v16h b2 = load_b_rm(v, 64, k0, 32);
    v16h b3 = load_b_rm(v, 64, k0, 48);
    c0 = __builtin_amdgcn_wmma_f32_16x16x32_f16(false, a, false, b0, (short)0, c0, false, false);
    c1 = __builtin_amdgcn_wmma_f32_16x16x32_f16(false, a, false, b1, (short)0, c1, false, false);
    c2 = __builtin_amdgcn_wmma_f32_16x16x32_f16(false, a, false, b2, (short)0, c2, false, false);
    c3 = __builtin_amdgcn_wmma_f32_16x16x32_f16(false, a, false, b3, (short)0, c3, false, false);
  }
  const int b = bh >> 3, h = bh & 7;
  v8f cc[4] = {c0, c1, c2, c3};
#pragma unroll
  for (int t = 0; t < 4; ++t) {
    const int d = t * 16 + (lane & 15);
#pragma unroll
    for (int r = 0; r < 8; ++r) {
      const int qrow = tm * 16 + r + ((lane & 16) ? 8 : 0);
      AO[(((size_t)(b * 512 + qrow)) * 8 + h) * 64 + d] = cc[t][r];
    }
  }
}

// ---- Kernel 7: out = AO(2048x512) @ Wo(512x512) + bo (WMMA, 16x64 strip) ----
__global__ void oproj_kernel(const float* __restrict__ AO, const float* __restrict__ Wo,
                             const float* __restrict__ bo, float* __restrict__ Out) {
  const int wave = (blockIdx.x * blockDim.x + threadIdx.x) >> 5;  // 1024 strips
  const int lane = threadIdx.x & 31;
  const int tm = wave >> 3, ts = wave & 7;
  v8f c0 = {}, c1 = {}, c2 = {}, c3 = {};
  for (int k0 = 0; k0 < 512; k0 += 32) {
    v16h a = load_a_rm(AO, 512, tm * 16, k0);
    __builtin_prefetch(AO + (size_t)(tm * 16 + (lane & 15)) * 512 + k0 + 32, 0, 1);
    v16h b0 = load_b_rm(Wo, 512, k0, ts * 64);
    v16h b1 = load_b_rm(Wo, 512, k0, ts * 64 + 16);
    v16h b2 = load_b_rm(Wo, 512, k0, ts * 64 + 32);
    v16h b3 = load_b_rm(Wo, 512, k0, ts * 64 + 48);
    c0 = __builtin_amdgcn_wmma_f32_16x16x32_f16(false, a, false, b0, (short)0, c0, false, false);
    c1 = __builtin_amdgcn_wmma_f32_16x16x32_f16(false, a, false, b1, (short)0, c1, false, false);
    c2 = __builtin_amdgcn_wmma_f32_16x16x32_f16(false, a, false, b2, (short)0, c2, false, false);
    c3 = __builtin_amdgcn_wmma_f32_16x16x32_f16(false, a, false, b3, (short)0, c3, false, false);
  }
  v8f cc[4] = {c0, c1, c2, c3};
#pragma unroll
  for (int t = 0; t < 4; ++t) {
    const int j = ts * 64 + t * 16 + (lane & 15);
    const float bj = bo[j];
#pragma unroll
    for (int r = 0; r < 8; ++r) {
      const int m = tm * 16 + r + ((lane & 16) ? 8 : 0);
      Out[(size_t)m * 512 + j] = cc[t][r] + bj;
    }
  }
}

extern "C" void kernel_launch(void* const* d_in, const int* in_sizes, int n_in,
                              void* d_out, int out_size, void* d_ws, size_t ws_size,
                              hipStream_t stream) {
  (void)in_sizes; (void)n_in; (void)out_size; (void)ws_size;
  const float* queries = (const float*)d_in[0];
  const float* keys    = (const float*)d_in[1];
  const float* values  = (const float*)d_in[2];
  const float* f_g     = (const float*)d_in[3];
  const float* Wq = (const float*)d_in[4];
  const float* bq = (const float*)d_in[5];
  const float* Wk = (const float*)d_in[6];
  const float* bk = (const float*)d_in[7];
  const float* Wv = (const float*)d_in[8];
  const float* bv = (const float*)d_in[9];
  const float* Wo = (const float*)d_in[10];
  const float* bo = (const float*)d_in[11];
  const float* Wg = (const float*)d_in[12];
  const float* bg = (const float*)d_in[13];

  char* ws = (char*)d_ws;
  const size_t SZ_BOXES = (size_t)BATCH * NTOT * 4 * sizeof(float);             // 32 KB
  const size_t SZ_GB    = (size_t)BATCH * HEADS * NTOT * NTOT * sizeof(float);  // 32 MB
  const size_t SZ_QKV   = (size_t)BATCH * HEADS * NTOT * DK * sizeof(float);    // 4 MB each
  float* boxes = (float*)(ws);
  float* gb    = (float*)(ws + SZ_BOXES);
  float* qb    = (float*)(ws + SZ_BOXES + SZ_GB);
  float* kb    = (float*)(ws + SZ_BOXES + SZ_GB + SZ_QKV);
  float* vb    = (float*)(ws + SZ_BOXES + SZ_GB + 2 * SZ_QKV);
  float* ao    = (float*)(ws + SZ_BOXES + SZ_GB + 3 * SZ_QKV);

  boxes_kernel<<<8, 256, 0, stream>>>(f_g, boxes);
  geom_bias_kernel<<<4096, 256, 0, stream>>>(boxes, Wg, bg, gb);
  proj_kernel<<<128, 256, 0, stream>>>(queries, Wq, bq, qb);
  proj_kernel<<<128, 256, 0, stream>>>(keys, Wk, bk, kb);
  proj_kernel<<<128, 256, 0, stream>>>(values, Wv, bv, vb);
  qk_bias_kernel<<<1024, 256, 0, stream>>>(qb, kb, gb);
  softmax_kernel<<<16384, 256, 0, stream>>>(gb);
  pv_kernel<<<128, 256, 0, stream>>>(gb, vb, ao);
  oproj_kernel<<<128, 256, 0, stream>>>(ao, Wo, bo, (float*)d_out);
}
